// TopKMoEAttention_15204184227981
// MI455X (gfx1250) — matrix operand order repaired and checked by reading, not compiled, not run on hardware
//
#include <hip/hip_runtime.h>
#include <hip/hip_bf16.h>

#define D_MODEL 768
#define NHEADS  12
#define DH      64
#define NEXP    4
#define SPAN    64
#define GH      128
#define BB      4
#define TT      4096
#define NSPAN   (TT / SPAN)            // 64
#define NSPANS  (BB * NSPAN)           // 256
#define SPANG   4                      // spans per MoE block
#define OUTN    ((size_t)BB * TT * D_MODEL)  // 12582912

typedef __attribute__((ext_vector_type(16))) __bf16 v16bf;
typedef __attribute__((ext_vector_type(8)))  float  v8f;
typedef __attribute__((ext_vector_type(4)))  unsigned int u32x4;
typedef __attribute__((ext_vector_type(8)))  unsigned int u32x8;

static __device__ __forceinline__ __bf16 f2bf(float f) { return (__bf16)f; }

union FragU { struct { u32x4 lo, hi; } u; v16bf v; };

// A-matrix 16x32 bf16 fragment (ISA 7.12.2): lane half 0 -> K=0..7 & 16..23,
// half 1 -> K=8..15 & 24..31. Two 16B chunks at +hf*8 and +hf*8+16 elements.
static __device__ __forceinline__ v16bf frag_a_ld(const __bf16* base, int row, int ld,
                                                  int koff, int hf) {
  const __bf16* p = base + row * ld + koff + hf * 8;
  FragU x;
  x.u.lo = *(const u32x4*)(p);
  x.u.hi = *(const u32x4*)(p + 16);
  return x.v;
}

// B-matrix 32x16 bf16 fragment: lanes 0-15 hold K=0..7 & 16..23, lanes 16-31 K=8..15 & 24..31.
static __device__ __forceinline__ v16bf frag_b_ld(const __bf16* base, int row, int ld,
                                                  int koff, int hf) {
  const __bf16* p = base + row * ld + koff + hf * 8;
  FragU x;
  x.u.lo = *(const u32x4*)(p);
  x.u.hi = *(const u32x4*)(p + 16);
  return x.v;
}

static __device__ __forceinline__ v8f wmma_bf16(v16bf a, v16bf b, v8f c) {
  return __builtin_amdgcn_wmma_f32_16x16x32_bf16(false, a, false, b, (short)0, c,
                                                 false, false);
}

// ---- Tensor Data Mover: 2D bf16 tile (tile_w x tile_h, row stride in elems)
// from global to LDS, rows packed contiguously in LDS. D# per ISA ch.8.
static __device__ __forceinline__ void tdm_load_bf16_2d(unsigned int lds_off,
                                                        const __bf16* gsrc,
                                                        unsigned int tile_w,
                                                        unsigned int tile_h,
                                                        unsigned int stride_elems) {
  unsigned long long ga = (unsigned long long)(size_t)gsrc;
  u32x4 g0;
  g0[0] = 1u;                                                   // count=1, user D#
  g0[1] = lds_off;                                              // lds_addr
  g0[2] = (unsigned int)ga;                                     // global_addr[31:0]
  g0[3] = ((unsigned int)(ga >> 32) & 0x01FFFFFFu) | (2u << 30);// addr[56:32]|type=2
  u32x8 g1;
  g1[0] = (1u << 16);                                           // data_size=1 (2B)
  g1[1] = (stride_elems & 0xFFFFu) << 16;                       // tensor_dim0 lo
  g1[2] = ((stride_elems >> 16) & 0xFFFFu) | ((tile_h & 0xFFFFu) << 16);
  g1[3] = (tile_w & 0xFFFFu) << 16;                             // tile_dim0
  g1[4] = (tile_h & 0xFFFFu);                                   // tile_dim1
  g1[5] = stride_elems;                                         // dim0_stride lo32
  g1[6] = 0u;
  g1[7] = 0u;
  asm volatile("tensor_load_to_lds %0, %1" :: "s"(g0), "s"(g1) : "memory");
}

static __device__ __forceinline__ void tdm_wait() {
  __builtin_amdgcn_s_wait_tensorcnt(0);
}

static __device__ __forceinline__ unsigned int lds_off_of(const void* p) {
  return (unsigned int)(size_t)p;  // LDS aperture: addr[31:0] = LDS offset
}

// ---------------------------------------------------------------- f32 -> bf16
__global__ void cvt_bf16_kernel(const float* __restrict__ in, __bf16* __restrict__ out,
                                int n) {
  int i0 = (blockIdx.x * 128 + threadIdx.x) * 8;
  if (i0 + 8 <= n) {
#pragma unroll
    for (int k = 0; k < 8; ++k) out[i0 + k] = f2bf(in[i0 + k]);
  } else {
    for (int i = i0; i < n; ++i) out[i] = f2bf(in[i]);
  }
}

// ------------------------------------------------------------------- gating
__global__ void gate_kernel(const float* __restrict__ x,
                            const float* __restrict__ gW1, const float* __restrict__ gb1,
                            const float* __restrict__ gW2, const float* __restrict__ gb2,
                            float* __restrict__ wout, float* __restrict__ entout) {
  __shared__ float pooled[D_MODEL];
  __shared__ float hbuf[GH];
  __shared__ float logits[NEXP];
  int span = blockIdx.x;
  int b = span >> 6, n = span & 63;
  int tid = threadIdx.x;

  for (int c = tid; c < D_MODEL; c += 128) {
    const float* base = x + ((size_t)b * TT + n * SPAN) * D_MODEL + c;
    float s = 0.f;
    for (int r = 0; r < SPAN; ++r) s += base[(size_t)r * D_MODEL];
    pooled[c] = s * (1.f / (float)SPAN);
  }
  __syncthreads();
  {
    float a = gb1[tid];
    const float* wrow = gW1 + (size_t)tid * D_MODEL;
    for (int c = 0; c < D_MODEL; ++c) a += pooled[c] * wrow[c];
    hbuf[tid] = fmaxf(a, 0.f);
  }
  __syncthreads();
  if (tid < NEXP) {
    float a = gb2[tid];
    const float* wrow = gW2 + tid * GH;
    for (int j = 0; j < GH; ++j) a += hbuf[j] * wrow[j];
    logits[tid] = a;
  }
  __syncthreads();
  if (tid == 0) {
    int i0 = 0;
    for (int e = 1; e < NEXP; ++e) if (logits[e] > logits[i0]) i0 = e;
    int i1 = -1;
    for (int e = 0; e < NEXP; ++e)
      if (e != i0 && (i1 < 0 || logits[e] > logits[i1])) i1 = e;
    float e1 = __expf(logits[i1] - logits[i0]);
    float p0 = 1.f / (1.f + e1);
    float p1 = e1 / (1.f + e1);
    float w4[NEXP] = {0.f, 0.f, 0.f, 0.f};
    w4[i0] = p0; w4[i1] = p1;
    float ent = 0.f;
#pragma unroll
    for (int e = 0; e < NEXP; ++e) {
      wout[span * NEXP + e] = w4[e];
      ent += w4[e] * __logf(fmaxf(w4[e], 1e-9f));
    }
    entout[span] = ent;
  }
}

__global__ void entropy_kernel(const float* __restrict__ entin, float* __restrict__ out) {
  __shared__ float s[NSPANS];
  s[threadIdx.x] = entin[threadIdx.x];
  __syncthreads();
  for (int st = NSPANS / 2; st > 0; st >>= 1) {
    if (threadIdx.x < st) s[threadIdx.x] += s[threadIdx.x + st];
    __syncthreads();
  }
  if (threadIdx.x == 0) out[0] = -s[0] * (1.f / (float)NSPANS);
}

// --------------------------------------------- MoE QKV: gate-combined GEMM
// grid = 3 * 12 * (NSPANS/SPANG) blocks; 256 threads = 8 waves.
// Wave w handles M-strip (w&3) of span pair (w>>2): 2 spans x 4 tiles = 64
// accumulator VGPRs per wave (no spills), while the 4-expert f32 weight tile
// is staged once per block per K-step (4x less L2 traffic on W).
__global__ void moe_qkv_kernel(const __bf16* __restrict__ xbf,
                               const float* __restrict__ Wq, const float* __restrict__ bq,
                               const float* __restrict__ Wk, const float* __restrict__ bk,
                               const float* __restrict__ Wv, const float* __restrict__ bv,
                               const float* __restrict__ wgate,
                               __bf16* __restrict__ qb, __bf16* __restrict__ kb,
                               __bf16* __restrict__ vb) {
  __shared__ float  Wf[NEXP * 64 * 32];       // 32KB expert weight f32 tile
  __shared__ __bf16 Aall[SPANG][64 * 32];     // 16KB x tiles, [s][k]
  __shared__ __bf16 Beff[SPANG][64 * 32];     // 16KB combined W_eff, [o][k]

  int bid = blockIdx.x;
  int ot = bid % 12;
  int sg = (bid / 12) % (NSPANS / SPANG);
  int which = bid / (12 * (NSPANS / SPANG));
  int span0 = sg * SPANG;

  const float* W    = (which == 0) ? Wq : (which == 1) ? Wk : Wv;
  const float* bias = (which == 0) ? bq : (which == 1) ? bk : bv;
  __bf16* outp      = (which == 0) ? qb : (which == 1) ? kb : vb;

  int tid = threadIdx.x;
  int lane = tid & 31, hf = lane >> 4, l16 = lane & 15;
  int wave = tid >> 5;            // 0..7
  int mo = (wave & 3) * 16;       // M strip
  int gbase = (wave >> 2) * 2;    // span pair: {0,1} or {2,3}

  float wgt[SPANG][NEXP];
#pragma unroll
  for (int g = 0; g < SPANG; ++g)
#pragma unroll
    for (int e = 0; e < NEXP; ++e) wgt[g][e] = wgate[(span0 + g) * NEXP + e];

  v8f acc[2][4];
#pragma unroll
  for (int g = 0; g < 2; ++g)
#pragma unroll
    for (int j = 0; j < 4; ++j)
      acc[g][j] = (v8f){0.f, 0.f, 0.f, 0.f, 0.f, 0.f, 0.f, 0.f};

  for (int kt = 0; kt < D_MODEL / 32; ++kt) {
    __syncthreads();  // previous iteration's LDS reads complete
    // TDM: async-load the 4 x-tiles (32 wide x 64 rows, stride 768) to LDS
    if (tid < 32) {
#pragma unroll
      for (int g = 0; g < SPANG; ++g) {
        int sp = span0 + g;
        const __bf16* src = xbf + ((size_t)(sp >> 6) * TT + (sp & 63) * SPAN) * D_MODEL;
        tdm_load_bf16_2d(lds_off_of(&Aall[g][0]), src + kt * 32, 32, 64, D_MODEL);
      }
    }
    // stage expert f32 weight tile: 4 x 64 rows x 32 f32 (16B chunks)
    for (int i = tid; i < NEXP * 64 * 8; i += 256) {
      int e = i >> 9, rem = i & 511, row = rem >> 3, c = rem & 7;
      ((u32x4*)Wf)[i] = *(const u32x4*)(W + ((size_t)e * D_MODEL + ot * 64 + row) *
                                              D_MODEL + kt * 32 + c * 4);
    }
    __syncthreads();  // Wf visible to all
    // combine per-span W_eff (reuses the 4 f32 LDS reads across 4 spans)
    for (int i = tid; i < 64 * 32; i += 256) {
      float w0 = Wf[0 * 2048 + i], w1 = Wf[1 * 2048 + i];
      float w2 = Wf[2 * 2048 + i], w3 = Wf[3 * 2048 + i];
#pragma unroll
      for (int g = 0; g < SPANG; ++g)
        Beff[g][i] = f2bf(wgt[g][0] * w0 + wgt[g][1] * w1 +
                          wgt[g][2] * w2 + wgt[g][3] * w3);
    }
    if (tid < 32) tdm_wait();
    __syncthreads();  // Beff + TDM A tiles visible

#pragma unroll
    for (int gg = 0; gg < 2; ++gg) {
      int g = gbase + gg;
      v16bf a  = frag_a_ld(&Aall[g][0], mo + l16, 32, 0, hf);
      v16bf b0 = frag_b_ld(&Beff[g][0], 0 * 16 + l16, 32, 0, hf);
      v16bf b1 = frag_b_ld(&Beff[g][0], 1 * 16 + l16, 32, 0, hf);
      v16bf b2 = frag_b_ld(&Beff[g][0], 2 * 16 + l16, 32, 0, hf);
      v16bf b3 = frag_b_ld(&Beff[g][0], 3 * 16 + l16, 32, 0, hf);
      acc[gg][0] = wmma_bf16(a, b0, acc[gg][0]);
      acc[gg][1] = wmma_bf16(a, b1, acc[gg][1]);
      acc[gg][2] = wmma_bf16(a, b2, acc[gg][2]);
      acc[gg][3] = wmma_bf16(a, b3, acc[gg][3]);
    }
  }

  // epilogue: gate-combined bias, convert, store bf16 (B,T,C)
#pragma unroll
  for (int gg = 0; gg < 2; ++gg) {
    int g = gbase + gg;
    int sp = span0 + g;
    __bf16* orow = outp + ((size_t)(sp >> 6) * TT + (sp & 63) * SPAN) * D_MODEL;
#pragma unroll
    for (int j = 0; j < 4; ++j) {
      int o = ot * 64 + j * 16 + l16;
      float be = 0.f;
#pragma unroll
      for (int e = 0; e < NEXP; ++e) be += wgt[g][e] * bias[e * D_MODEL + o];
#pragma unroll
      for (int r = 0; r < 8; ++r) {
        int s = mo + r + 8 * hf;
        orow[(size_t)s * D_MODEL + o] = f2bf(acc[gg][j][r] + be);
      }
    }
  }
}

// ------------------------------------------------- per-span attention (WMMA)
__global__ void attn_kernel(const __bf16* __restrict__ qbf, const __bf16* __restrict__ kbf,
                            const __bf16* __restrict__ vbf, __bf16* __restrict__ obf) {
  __shared__ __bf16 Qs[64 * 64];
  __shared__ __bf16 Ks[64 * 64];
  __shared__ float  Sc[64 * 64];
  __shared__ __bf16 Ps[64 * 64];
  __shared__ __bf16 Vt[64 * 64];   // transposed: [d][t]

  int h = blockIdx.x % NHEADS;
  int span = blockIdx.x / NHEADS;
  int b = span >> 6, n = span & 63;

  int tid = threadIdx.x;
  int lane = tid & 31, hf = lane >> 4, l16 = lane & 15;
  int mo = (tid >> 5) * 16;

  size_t rowbase = ((size_t)b * TT + n * SPAN) * D_MODEL + h * DH;

  if (tid < 32) {  // TDM: Q and K 64x64 tiles, stride 768
    tdm_load_bf16_2d(lds_off_of(&Qs[0]), qbf + rowbase, 64, 64, D_MODEL);
    tdm_load_bf16_2d(lds_off_of(&Ks[0]), kbf + rowbase, 64, 64, D_MODEL);
  }
  for (int i = tid; i < 64 * 64; i += 128) {  // transpose V by hand
    int t = i >> 6, d = i & 63;
    Vt[d * 64 + t] = vbf[rowbase + (size_t)t * D_MODEL + d];
  }
  if (tid < 32) tdm_wait();
  __syncthreads();

  // scores = Q K^T / sqrt(Dh)
  v8f acc[4];
#pragma unroll
  for (int j = 0; j < 4; ++j) acc[j] = (v8f){0.f, 0.f, 0.f, 0.f, 0.f, 0.f, 0.f, 0.f};
#pragma unroll
  for (int kt = 0; kt < 2; ++kt) {
    v16bf a  = frag_a_ld(Qs, mo + l16, 64, kt * 32, hf);
    v16bf b0 = frag_b_ld(Ks, 0 * 16 + l16, 64, kt * 32, hf);
    v16bf b1 = frag_b_ld(Ks, 1 * 16 + l16, 64, kt * 32, hf);
    v16bf b2 = frag_b_ld(Ks, 2 * 16 + l16, 64, kt * 32, hf);
    v16bf b3 = frag_b_ld(Ks, 3 * 16 + l16, 64, kt * 32, hf);
    acc[0] = wmma_bf16(a, b0, acc[0]);
    acc[1] = wmma_bf16(a, b1, acc[1]);
    acc[2] = wmma_bf16(a, b2, acc[2]);
    acc[3] = wmma_bf16(a, b3, acc[3]);
  }
#pragma unroll
  for (int j = 0; j < 4; ++j)
#pragma unroll
    for (int r = 0; r < 8; ++r)
      Sc[(mo + r + 8 * hf) * 64 + j * 16 + l16] = acc[j][r] * 0.125f;
  __syncthreads();

  if (tid < 64) {  // row softmax, probs -> bf16
    float m = -3.4e38f;
    for (int t = 0; t < 64; ++t) m = fmaxf(m, Sc[tid * 64 + t]);
    float s = 0.f;
    for (int t = 0; t < 64; ++t) {
      float e = __expf(Sc[tid * 64 + t] - m);
      Sc[tid * 64 + t] = e;
      s += e;
    }
    float inv = 1.f / s;
    for (int t = 0; t < 64; ++t) Ps[tid * 64 + t] = f2bf(Sc[tid * 64 + t] * inv);
  }
  __syncthreads();

  // out = P V
  v8f oacc[4];
#pragma unroll
  for (int j = 0; j < 4; ++j) oacc[j] = (v8f){0.f, 0.f, 0.f, 0.f, 0.f, 0.f, 0.f, 0.f};
#pragma unroll
  for (int kt = 0; kt < 2; ++kt) {
    v16bf a  = frag_a_ld(Ps, mo + l16, 64, kt * 32, hf);
    v16bf b0 = frag_b_ld(Vt, 0 * 16 + l16, 64, kt * 32, hf);
    v16bf b1 = frag_b_ld(Vt, 1 * 16 + l16, 64, kt * 32, hf);
    v16bf b2 = frag_b_ld(Vt, 2 * 16 + l16, 64, kt * 32, hf);
    v16bf b3 = frag_b_ld(Vt, 3 * 16 + l16, 64, kt * 32, hf);
    oacc[0] = wmma_bf16(a, b0, oacc[0]);
    oacc[1] = wmma_bf16(a, b1, oacc[1]);
    oacc[2] = wmma_bf16(a, b2, oacc[2]);
    oacc[3] = wmma_bf16(a, b3, oacc[3]);
  }
#pragma unroll
  for (int j = 0; j < 4; ++j)
#pragma unroll
    for (int r = 0; r < 8; ++r) {
      int s = mo + r + 8 * hf;
      obf[rowbase + (size_t)s * D_MODEL + j * 16 + l16] = f2bf(oacc[j][r]);
    }
}

// --------------------------------------------------- output projection GEMM
__global__ void outproj_kernel(const __bf16* __restrict__ abf,
                               const __bf16* __restrict__ wobf,
                               const float* __restrict__ bo, float* __restrict__ out) {
  __shared__ __bf16 Alds[64 * 32];
  __shared__ __bf16 Blds[64 * 32];

  int ot = blockIdx.x % 12;
  int rt = blockIdx.x / 12;
  int row0 = rt * 64;

  int tid = threadIdx.x;
  int lane = tid & 31, hf = lane >> 4, l16 = lane & 15;
  int mo = (tid >> 5) * 16;

  v8f acc[4];
#pragma unroll
  for (int j = 0; j < 4; ++j) acc[j] = (v8f){0.f, 0.f, 0.f, 0.f, 0.f, 0.f, 0.f, 0.f};

  for (int kt = 0; kt < D_MODEL / 32; ++kt) {
    __syncthreads();
    if (tid < 32) {
      tdm_load_bf16_2d(lds_off_of(&Alds[0]), abf + (size_t)row0 * D_MODEL + kt * 32,
                       32, 64, D_MODEL);
      tdm_load_bf16_2d(lds_off_of(&Blds[0]), wobf + (size_t)(ot * 64) * D_MODEL + kt * 32,
                       32, 64, D_MODEL);
      tdm_wait();
    }
    __syncthreads();

    v16bf a  = frag_a_ld(Alds, mo + l16, 32, 0, hf);
    v16bf b0 = frag_b_ld(Blds, 0 * 16 + l16, 32, 0, hf);
    v16bf b1 = frag_b_ld(Blds, 1 * 16 + l16, 32, 0, hf);
    v16bf b2 = frag_b_ld(Blds, 2 * 16 + l16, 32, 0, hf);
    v16bf b3 = frag_b_ld(Blds, 3 * 16 + l16, 32, 0, hf);
    acc[0] = wmma_bf16(a, b0, acc[0]);
    acc[1] = wmma_bf16(a, b1, acc[1]);
    acc[2] = wmma_bf16(a, b2, acc[2]);
    acc[3] = wmma_bf16(a, b3, acc[3]);
  }

#pragma unroll
  for (int j = 0; j < 4; ++j) {
    int o = ot * 64 + j * 16 + l16;
    float be = bo[o];
#pragma unroll
    for (int r = 0; r < 8; ++r) {
      int s = mo + r + 8 * hf;
      out[((size_t)(row0 + s)) * D_MODEL + o] = acc[j][r] + be;
    }
  }
}

// ---------------------------------------------------------------- launcher
static inline size_t align256(size_t x) { return (x + 255) & ~(size_t)255; }

extern "C" void kernel_launch(void* const* d_in, const int* in_sizes, int n_in,
                              void* d_out, int out_size, void* d_ws, size_t ws_size,
                              hipStream_t stream) {
  const float* x   = (const float*)d_in[0];
  const float* Wq  = (const float*)d_in[1];
  const float* bq  = (const float*)d_in[2];
  const float* Wk  = (const float*)d_in[3];
  const float* bk  = (const float*)d_in[4];
  const float* Wv  = (const float*)d_in[5];
  const float* bv  = (const float*)d_in[6];
  const float* gW1 = (const float*)d_in[7];
  const float* gb1 = (const float*)d_in[8];
  const float* gW2 = (const float*)d_in[9];
  const float* gb2 = (const float*)d_in[10];
  const float* Wo  = (const float*)d_in[11];
  const float* bo  = (const float*)d_in[12];
  float* out = (float*)d_out;

  char* p = (char*)d_ws;
  float* ws_w   = (float*)p;  p += align256(NSPANS * NEXP * sizeof(float));
  float* ws_ent = (float*)p;  p += align256(NSPANS * sizeof(float));
  __bf16* xbf   = (__bf16*)p; p += align256(OUTN * sizeof(__bf16));
  __bf16* qb    = (__bf16*)p; p += align256(OUTN * sizeof(__bf16));
  __bf16* kb    = (__bf16*)p; p += align256(OUTN * sizeof(__bf16));
  __bf16* vb    = (__bf16*)p; p += align256(OUTN * sizeof(__bf16));
  __bf16* ab    = (__bf16*)p; p += align256(OUTN * sizeof(__bf16));
  __bf16* wob   = (__bf16*)p; p += align256((size_t)D_MODEL * D_MODEL * sizeof(__bf16));
  (void)ws_size; (void)n_in; (void)in_sizes; (void)out_size;

  int nx = (int)OUTN;
  cvt_bf16_kernel<<<(nx + 1023) / 1024, 128, 0, stream>>>(x, xbf, nx);
  int nwo = D_MODEL * D_MODEL;
  cvt_bf16_kernel<<<(nwo + 1023) / 1024, 128, 0, stream>>>(Wo, wob, nwo);

  gate_kernel<<<NSPANS, 128, 0, stream>>>(x, gW1, gb1, gW2, gb2, ws_w, ws_ent);
  entropy_kernel<<<1, NSPANS, 0, stream>>>(ws_ent, out + OUTN);

  moe_qkv_kernel<<<3 * 12 * (NSPANS / SPANG), 256, 0, stream>>>(xbf, Wq, bq, Wk, bk,
                                                                Wv, bv, ws_w, qb, kb, vb);
  attn_kernel<<<NSPANS * NHEADS, 128, 0, stream>>>(qb, kb, vb, ab);
  outproj_kernel<<<(BB * TT / 64) * 12, 128, 0, stream>>>(ab, wob, bo, out);
}
